// KANLayer_34351148434027
// MI455X (gfx1250) — matrix operand order repaired and checked by reading, not compiled, not run on hardware
//
#include <hip/hip_runtime.h>
#include <hip/hip_bf16.h>
#include <stdint.h>

typedef __attribute__((ext_vector_type(16))) __bf16 v16bf;
typedef __attribute__((ext_vector_type(8)))  float  v8f;

#define IN_DIM   512
#define OUT_DIM  512
#define NBATCH   256
#define KDEG     3
#define NGRID    23          // NUM + 2K + 1
#define GBASIS   19          // NUM + K
#define GP       20          // 19 basis + 1 silu slot
#define KP       (IN_DIM * GP)   // 10240

// round-to-nearest-even f32 -> bf16
static __device__ __forceinline__ unsigned short f2bf(float f) {
    unsigned u = __builtin_bit_cast(unsigned, f);
    u += 0x7FFFu + ((u >> 16) & 1u);
    return (unsigned short)(u >> 16);
}

// Cox-de-Boor cubic basis: 23 grid pts -> 19 basis values, plus silu(x)
static __device__ __forceinline__ void kan_features(float xv, const float* __restrict__ grow,
                                                    float* __restrict__ feat /*20*/) {
    float gv[NGRID];
#pragma unroll
    for (int j = 0; j < NGRID; ++j) gv[j] = grow[j];
    float Bv[NGRID - 1];
#pragma unroll
    for (int j = 0; j < NGRID - 1; ++j)
        Bv[j] = (xv >= gv[j] && xv < gv[j + 1]) ? 1.0f : 0.0f;
#pragma unroll
    for (int kk = 1; kk <= KDEG; ++kk) {
#pragma unroll
        for (int j = 0; j < NGRID - 1 - kk; ++j) {
            // in-place forward update: uses old Bv[j], old Bv[j+1]
            Bv[j] = (xv - gv[j]) / (gv[j + kk] - gv[j]) * Bv[j]
                  + (gv[j + kk + 1] - xv) / (gv[j + kk + 1] - gv[j + 1]) * Bv[j + 1];
        }
    }
#pragma unroll
    for (int j = 0; j < GBASIS; ++j) feat[j] = Bv[j];
    feat[GBASIS] = xv / (1.0f + __expf(-xv));   // silu
}

// A[b, i*20+g] in bf16, row length KP
__global__ void build_A(const float* __restrict__ x, const float* __restrict__ grid,
                        unsigned short* __restrict__ A) {
    int tid = blockIdx.x * blockDim.x + threadIdx.x;
    if (tid >= NBATCH * IN_DIM) return;
    int i = tid & (IN_DIM - 1);
    int b = tid >> 9;
    float feat[GP];
    kan_features(x[(size_t)b * IN_DIM + i], grid + (size_t)i * NGRID, feat);
    unsigned short* dst = A + (size_t)b * KP + (size_t)i * GP;
    uint32_t* d32 = (uint32_t*)dst;   // 40B chunk, 8B aligned
#pragma unroll
    for (int j = 0; j < GP / 2; ++j) {
        uint32_t lo = f2bf(feat[2 * j]);
        uint32_t hi = f2bf(feat[2 * j + 1]);
        d32[j] = lo | (hi << 16);
    }
}

// Wt[o, i*20+g] in bf16 (o-major, row length KP): coef*scale*mask fused; slot 19 = mask
__global__ void build_W(const float* __restrict__ coef, const float* __restrict__ ssp,
                        const float* __restrict__ mask, unsigned short* __restrict__ Wt) {
    int tid = blockIdx.x * blockDim.x + threadIdx.x;
    if (tid >= IN_DIM * OUT_DIM) return;
    int o = tid & (OUT_DIM - 1);
    int i = tid >> 9;
    float m  = mask[(size_t)i * OUT_DIM + o];
    float sm = ssp [(size_t)i * OUT_DIM + o] * m;
    const float* c = coef + ((size_t)i * OUT_DIM + o) * GBASIS;
    float w[GP];
#pragma unroll
    for (int j = 0; j < GBASIS; ++j) w[j] = c[j] * sm;
    w[GBASIS] = m;
    unsigned short* dst = Wt + (size_t)o * KP + (size_t)i * GP;
    uint32_t* d32 = (uint32_t*)dst;
#pragma unroll
    for (int j = 0; j < GP / 2; ++j) {
        uint32_t lo = f2bf(w[2 * j]);
        uint32_t hi = f2bf(w[2 * j + 1]);
        d32[j] = lo | (hi << 16);
    }
}

union FragBF {
    uint4  q[2];
    v16bf  v;
};

// One wave per 16x32 output tile: the A fragment is loaded once per k-step and
// fed to TWO v_wmma_f32_16x16x32_bf16 (two 16x16 accumulators) -> 6 b128 loads
// per 2 WMMA instead of 8. 256 waves total (64 blocks x 4 waves).
__global__ __launch_bounds__(128) void gemm_wmma(const unsigned short* __restrict__ A,
                                                 const unsigned short* __restrict__ Wt,
                                                 float* __restrict__ out) {
    const int lane = threadIdx.x & 31;
    const int wave = (blockIdx.x * (blockDim.x >> 5)) + (threadIdx.x >> 5);
    const int ob   = wave & 15;        // 16 super-tiles along OUT (512/32)
    const int bb   = wave >> 4;        // 16 tiles along BATCH (256/16)
    const int m    = lane & 15;
    const int h    = lane >> 4;

    // A fragment (16x32 bf16): lane holds row (bb*16+m), K chunks [8h..8h+7], [16+8h..23+8h]
    const unsigned short* aptr  = A  + (size_t)(bb * 16 + m) * KP + 8 * h;
    // B fragments (32x16 bf16): lane holds col, K chunks [8h..8h+7], [16+8h..23+8h]
    const unsigned short* bptr0 = Wt + (size_t)(ob * 32 + m) * KP + 8 * h;
    const unsigned short* bptr1 = bptr0 + (size_t)16 * KP;   // cols +16

    v8f c0 = {0.0f, 0.0f, 0.0f, 0.0f, 0.0f, 0.0f, 0.0f, 0.0f};
    v8f c1 = {0.0f, 0.0f, 0.0f, 0.0f, 0.0f, 0.0f, 0.0f, 0.0f};

#pragma unroll 2
    for (int k0 = 0; k0 < KP; k0 += 32) {
        FragBF fa, fb0, fb1;
        fa.q[0]  = *(const uint4*)(aptr  + k0);        // K = k0 + 8h    .. +7
        fa.q[1]  = *(const uint4*)(aptr  + k0 + 16);   // K = k0 + 16+8h .. +7
        fb0.q[0] = *(const uint4*)(bptr0 + k0);        // K = k0 + 8h    .. +7
        fb0.q[1] = *(const uint4*)(bptr0 + k0 + 16);   // K = k0 + 16+8h .. +7
        fb1.q[0] = *(const uint4*)(bptr1 + k0);
        fb1.q[1] = *(const uint4*)(bptr1 + k0 + 16);
        c0 = __builtin_amdgcn_wmma_f32_16x16x32_bf16(
                false, fa.v, false, fb0.v, (short)0, c0, false, false);
        c1 = __builtin_amdgcn_wmma_f32_16x16x32_bf16(
                false, fa.v, false, fb1.v, (short)0, c1, false, false);
    }

    // C/D layout: VGPR r, lanes 0-15 -> M=r, N=lane; lanes 16-31 -> M=r+8, N=lane-16
    float* obase = out + (size_t)(bb * 16 + 8 * h) * OUT_DIM + ob * 32 + m;
#pragma unroll
    for (int r = 0; r < 8; ++r) {
        obase[(size_t)r * OUT_DIM]      = c0[r];
        obase[(size_t)r * OUT_DIM + 16] = c1[r];
    }
}

// LDS fallback if workspace is too small (correctness path, not perf)
__global__ __launch_bounds__(256) void kan_fallback(const float* __restrict__ x,
                                                    const float* __restrict__ grid,
                                                    const float* __restrict__ coef,
                                                    const float* __restrict__ ssp,
                                                    const float* __restrict__ mask,
                                                    float* __restrict__ out) {
    __shared__ float sA[IN_DIM * GP];   // 40 KB (WGP has 320 KB LDS)
    int b = blockIdx.x;
    for (int i = threadIdx.x; i < IN_DIM; i += blockDim.x) {
        float feat[GP];
        kan_features(x[(size_t)b * IN_DIM + i], grid + (size_t)i * NGRID, feat);
#pragma unroll
        for (int j = 0; j < GP; ++j) sA[i * GP + j] = feat[j];
    }
    __syncthreads();
    for (int o = threadIdx.x; o < OUT_DIM; o += blockDim.x) {
        float acc = 0.0f;
        for (int i = 0; i < IN_DIM; ++i) {
            float mm = mask[(size_t)i * OUT_DIM + o];
            float ss = ssp [(size_t)i * OUT_DIM + o];
            const float* c = coef + ((size_t)i * OUT_DIM + o) * GBASIS;
            float y = 0.0f;
#pragma unroll
            for (int g = 0; g < GBASIS; ++g) y += sA[i * GP + g] * c[g];
            acc += (sA[i * GP + GBASIS] + ss * y) * mm;
        }
        out[(size_t)b * OUT_DIM + o] = acc;
    }
}

extern "C" void kernel_launch(void* const* d_in, const int* in_sizes, int n_in,
                              void* d_out, int out_size, void* d_ws, size_t ws_size,
                              hipStream_t stream) {
    const float* x    = (const float*)d_in[0];
    const float* grid = (const float*)d_in[1];
    const float* coef = (const float*)d_in[2];
    const float* ssp  = (const float*)d_in[3];
    const float* mask = (const float*)d_in[4];
    float* out = (float*)d_out;

    const size_t bytesA = (size_t)NBATCH * KP * sizeof(unsigned short);   // ~5 MB
    const size_t bytesW = (size_t)OUT_DIM * KP * sizeof(unsigned short);  // ~10 MB

    if (ws_size >= bytesA + bytesW) {
        unsigned short* A  = (unsigned short*)d_ws;
        unsigned short* Wt = (unsigned short*)((char*)d_ws + bytesA);
        build_A<<<(NBATCH * IN_DIM) / 256, 256, 0, stream>>>(x, grid, A);
        build_W<<<(IN_DIM * OUT_DIM) / 256, 256, 0, stream>>>(coef, ssp, mask, Wt);
        // 256 waves of 16x32 tiles; 4 waves (128 threads) per block -> 64 blocks
        gemm_wmma<<<64, 128, 0, stream>>>(A, Wt, out);
    } else {
        kan_fallback<<<NBATCH, 256, 0, stream>>>(x, grid, coef, ssp, mask, out);
    }
}
